// GCNLSTMPolicy_39341900431539
// MI455X (gfx1250) — matrix operand order repaired and checked by reading, not compiled, not run on hardware
//
#include <hip/hip_runtime.h>
#include <math.h>

// ---------------------------------------------------------------------------
// GCN-LSTM policy step on MI455X (gfx1250, wave32, WMMA).
// All GEMMs use v_wmma_f32_16x16x32_f16 (f32 accumulate). Workload is
// HBM-bound on agents_obs (~54.5 MB of the ~67 MB total), so the encoders do
// exactly one pass over the node observations; everything downstream is fused.
// All nonlinearities use the gfx1250 hardware v_tanh_f32 transcendental.
// ---------------------------------------------------------------------------

typedef __attribute__((ext_vector_type(16))) _Float16 v16h;
typedef __attribute__((ext_vector_type(8)))  float    v8f;

#define HID 32
#define H3  96
#define NGATE 384

__device__ __forceinline__ v8f wmma_f16(v16h a, v16h b, v8f c) {
    // D = A(16x32 f16) * B(32x16 f16) + C(16x16 f32)
    return __builtin_amdgcn_wmma_f32_16x16x32_f16(
        /*neg_a=*/false, a, /*neg_b=*/false, b,
        /*c_mod=*/(short)0, c, /*reuse_a=*/false, /*reuse_b=*/false);
}

// gfx1250 hardware tanh (TRANS32). Builtin when available; inline asm fallback
// with a trailing v_nop to satisfy the TRANS RAW hazard rule (ISA 7.4).
__device__ __forceinline__ float fast_tanh(float x) {
#if __has_builtin(__builtin_amdgcn_tanhf)
    return __builtin_amdgcn_tanhf(x);
#else
    float y;
    asm volatile("v_tanh_f32 %0, %1\n\tv_nop" : "=v"(y) : "v"(x));
    return y;
#endif
}
__device__ __forceinline__ float fast_sigmoid(float x) {
    return 0.5f * fast_tanh(0.5f * x) + 0.5f;   // exact identity, one TRANS op
}

// Build 16x32 f16 A-fragment from LDS (row-major, STRIDE halves per row).
// Layout: lanes 0-15 M=lane, lanes 16-31 M=lane-16; lanes 0-15 hold K=0-15,
// lanes 16-31 K=16-31; VGPR v holds K = 16*(lane>=16) + 2v + {0,1}.
template<int STRIDE>
__device__ __forceinline__ v16h load_A_frag(const _Float16* base, int lane, int kbase) {
    union { v16h vec; unsigned u[8]; } fa;
    const int M  = lane & 15;
    const int hi = lane >> 4;
    const _Float16* row = base + M * STRIDE + kbase;
#pragma unroll
    for (int vv = 0; vv < 8; ++vv) {
        int K0 = hi * 16 + vv * 2;
        fa.u[vv] = *(const unsigned*)(row + K0);   // two consecutive halves
    }
    return fa.vec;
}

// Build 32x16 f16 B-fragment from LDS weight tile stored [N][K] (STRIDE halves).
// Layout used here (same as load_A_frag): N = lane&15; lanes 0-15 hold K=0-15, lanes 16-31 K=16-31;
// VGPR v holds K = 16*(lane>=16) + 2v + {0,1}.
template<int STRIDE>
__device__ __forceinline__ v16h load_B_frag_lds(const _Float16* base, int lane, int n0, int kbase) {
    union { v16h vec; unsigned u[8]; } fb;
    const int N  = n0 + (lane & 15);
    const int hi = lane >> 4;
    const _Float16* row = base + N * STRIDE + kbase + hi * 16;
#pragma unroll
    for (int vv = 0; vv < 8; ++vv)
        fb.u[vv] = *(const unsigned*)(row + vv * 2);
    return fb.vec;
}

// B-fragment gathered from global f32 weights W[N][ldw] with zero K padding
// (used only for the tiny K=32/K=18 stage-2 GEMMs).
__device__ __forceinline__ v16h load_B_frag_g(const float* __restrict__ W, int ldw,
                                              int lane, int n0, int kbase, int kvalid) {
    union { v16h vec; _Float16 h[16]; } fb;
    const int N  = n0 + (lane & 15);
    const int hi = lane >> 4;
    const float* row = W + (size_t)N * ldw;
#pragma unroll
    for (int vv = 0; vv < 8; ++vv) {
        int K0 = kbase + hi * 16 + vv * 2;
        float a = (K0     < kvalid) ? row[K0]     : 0.f;
        float b = (K0 + 1 < kvalid) ? row[K0 + 1] : 0.f;
        fb.h[2 * vv]     = (_Float16)a;
        fb.h[2 * vv + 1] = (_Float16)b;
    }
    return fb.vec;
}

// B-fragment from global f32 weights, fully in-range: aligned float2 loads
// (global_load_b64) + pack-convert. ldw must be even (96 here).
__device__ __forceinline__ v16h load_B_frag_g2(const float* __restrict__ W, int ldw,
                                               int lane, int n0, int kbase) {
    union { v16h vec; _Float16 h[16]; } fb;
    const int N  = n0 + (lane & 15);
    const int hi = lane >> 4;
    const float* row = W + (size_t)N * ldw + kbase + hi * 16;
#pragma unroll
    for (int vv = 0; vv < 8; ++vv) {
        float2 p = *(const float2*)(row + vv * 2);   // 8B-aligned: even offsets
        fb.h[2 * vv]     = (_Float16)p.x;
        fb.h[2 * vv + 1] = (_Float16)p.y;
    }
    return fb.vec;
}

// ---------------------------------------------------------------------------
// Kernel 1/2: node encoder + mean pool.
//   mean_out[b,h] = mean_n tanh( x[b,n,:] @ W[h,:] + bias[h] ),  h < 32
// One block per batch element, NODES threads (NODES/32 waves, 2 M-tiles each).
// WMMA tiles 16(nodes) x 32(K padded) x 16(hid), hardware tanh on the C
// fragment, mean via LDS float atomics (ds_add_f32).
// ---------------------------------------------------------------------------
template<int NODES, int INDIM>
__global__ void __launch_bounds__(NODES)
encoder_mean_kernel(const float* __restrict__ x,     // [B, NODES, INDIM]
                    const float* __restrict__ W,     // [32, INDIM]
                    const float* __restrict__ bias,  // [32]
                    float* __restrict__ mean_out)    // [B, 32]
{
    __shared__ __attribute__((aligned(16))) _Float16 s_x[NODES][32];
    __shared__ __attribute__((aligned(16))) _Float16 s_w[32][32];
    __shared__ float s_sum[32];

    const int b   = blockIdx.x;
    const int tid = threadIdx.x;

    if (tid < 32) s_sum[tid] = 0.f;
    // zero-fill the padded K columns (2 halves per 32-bit store)
    for (int i = tid; i < NODES * 16; i += NODES)
        ((unsigned*)&s_x[0][0])[i] = 0u;
    for (int i = tid; i < 32 * 32; i += NODES) {
        int n = i >> 5, k = i & 31;
        s_w[n][k] = (_Float16)((k < INDIM) ? W[n * INDIM + k] : 0.f);
    }
    __syncthreads();
    // coalesced load of exactly NODES*INDIM observation floats
    const float* xb = x + (size_t)b * NODES * INDIM;
    for (int i = tid; i < NODES * INDIM; i += NODES) {
        int r = i / INDIM, k = i % INDIM;
        s_x[r][k] = (_Float16)xb[i];
    }
    __syncthreads();

    const int lane = tid & 31;
    const int wave = tid >> 5;
    const float bias0 = bias[lane & 15];
    const float bias1 = bias[16 + (lane & 15)];

    // Weight fragments are loop-invariant for the wave.
    v16h fb0 = load_B_frag_lds<32>(&s_w[0][0], lane, 0, 0);
    v16h fb1 = load_B_frag_lds<32>(&s_w[0][0], lane, 16, 0);

#pragma unroll
    for (int t = 0; t < 2; ++t) {
        const int m0 = wave * 32 + t * 16;
        v16h fa = load_A_frag<32>(&s_x[m0][0], lane, 0);
        v8f c0 = {0.f, 0.f, 0.f, 0.f, 0.f, 0.f, 0.f, 0.f};
        v8f c1 = {0.f, 0.f, 0.f, 0.f, 0.f, 0.f, 0.f, 0.f};
        c0 = wmma_f16(fa, fb0, c0);
        c1 = wmma_f16(fa, fb1, c1);
        float p0 = 0.f, p1 = 0.f;
#pragma unroll
        for (int r = 0; r < 8; ++r) {
            p0 += fast_tanh(c0[r] + bias0);
            p1 += fast_tanh(c1[r] + bias1);
        }
        atomicAdd(&s_sum[lane & 15], p0);        // ds_add_f32
        atomicAdd(&s_sum[16 + (lane & 15)], p1);
    }
    __syncthreads();
    if (tid < 32) mean_out[(size_t)b * 32 + tid] = s_sum[tid] * (1.0f / (float)NODES);
}

// ---------------------------------------------------------------------------
// Kernel 3: fused head. 16 batch rows per block (256 threads = 8 waves).
// Phase 1: waves 0-5 run the three K=32 stage-2 GEMMs (tanh for a/l) into the
//          concatenated f16 activation tile xcat[16][192] = [a|l|c|hx].
// Phase 2: gate GEMM gates = xcat @ [W_ih;W_hh]^T + b, K=192 (6 k-steps, each
//          k-step entirely inside W_ih or W_hh -> no address selects),
//          N=384 (24 n-tiles; 3 per wave), bias-preloaded accumulators.
// Phase 3: LSTM elementwise (hardware tanh/sigmoid). Phase 4: heads + argmax.
// ---------------------------------------------------------------------------
__global__ void __launch_bounds__(256)
head_kernel(const float* __restrict__ mean_a, const float* __restrict__ mean_l,
            const float* __restrict__ c_obs,                       // [B,18]
            const float* __restrict__ hx, const float* __restrict__ cx,  // [B,96]
            const float* __restrict__ W_agg_ag, const float* __restrict__ b_agg_ag,
            const float* __restrict__ W_agg_lm, const float* __restrict__ b_agg_lm,
            const float* __restrict__ W_c,      const float* __restrict__ b_c,
            const float* __restrict__ W_ih,     const float* __restrict__ b_ih,
            const float* __restrict__ W_hh,     const float* __restrict__ b_hh,
            const float* __restrict__ W_mov,    const float* __restrict__ b_mov,
            const float* __restrict__ W_int,    const float* __restrict__ b_int,
            float* __restrict__ out, int B)
{
    __shared__ __attribute__((aligned(16))) _Float16 s_in[16][96];    // xa|xl|xc(pad)
    __shared__ __attribute__((aligned(16))) _Float16 s_cat[16][192];  // [a|l|c|hx] f16
    __shared__ float s_gates[16][NGATE];
    __shared__ float s_hnew[16][H3];
    __shared__ float s_act[16][12];

    const int blk  = blockIdx.x;
    const int tid  = threadIdx.x;
    const int row0 = blk * 16;
    const int lane = tid & 31;
    const int wave = tid >> 5;

    const size_t OFF_AI = (size_t)4 * B;     // act_inter
    const size_t OFF_M1 = (size_t)12 * B;    // argmax mov
    const size_t OFF_M2 = (size_t)13 * B;    // argmax inter
    const size_t OFF_HX = (size_t)14 * B;    // hx_new
    const size_t OFF_CX = (size_t)110 * B;   // cx_new

    // warm the L2-resident gate weights (global_prefetch); 256 thr x 144 floats
    __builtin_prefetch(W_ih + (size_t)tid * 144, 0, 3);
    __builtin_prefetch(W_hh + (size_t)tid * 144, 0, 3);

    // --- phase 0: stage inputs as f16 ---
    for (int i = tid; i < 16 * 32; i += 256) {
        int m = i >> 5, k = i & 31;
        s_in[m][k]      = (_Float16)mean_a[(size_t)(row0 + m) * 32 + k];
        s_in[m][32 + k] = (_Float16)mean_l[(size_t)(row0 + m) * 32 + k];
        s_in[m][64 + k] = (_Float16)((k < 18) ? c_obs[(size_t)(row0 + m) * 18 + k] : 0.f);
    }
    for (int i = tid; i < 16 * 48; i += 256) {       // hx: float2 -> packed halves
        int m = i / 48, k2 = (i % 48) * 2;
        float2 p = *(const float2*)(hx + (size_t)(row0 + m) * H3 + k2);
        union { unsigned u; _Float16 h[2]; } pk;
        pk.h[0] = (_Float16)p.x; pk.h[1] = (_Float16)p.y;
        *(unsigned*)&s_cat[m][H3 + k2] = pk.u;
    }
    __syncthreads();

    // --- phase 1: a = tanh(xa W_agg_ag^T+b), l = tanh(...), c = linear ---
    if (wave < 6) {                       // wave-uniform branch: EXEC all-ones inside
        const int mat = wave >> 1;        // 0=a, 1=l, 2=c
        const int nt  = wave & 1;         // which 16-wide half of HID=32
        const float* Wm = (mat == 0) ? W_agg_ag : (mat == 1) ? W_agg_lm : W_c;
        const float* bm = (mat == 0) ? b_agg_ag : (mat == 1) ? b_agg_lm : b_c;
        const int kvalid = (mat == 2) ? 18 : 32;

        v16h fa = load_A_frag<96>(&s_in[0][mat * 32], lane, 0);
        v16h fb = load_B_frag_g(Wm, kvalid, lane, nt * 16, 0, kvalid);
        v8f  c  = {0.f, 0.f, 0.f, 0.f, 0.f, 0.f, 0.f, 0.f};
        c = wmma_f16(fa, fb, c);

        const float bn  = bm[nt * 16 + (lane & 15)];
        const int   hi8 = (lane >> 4) * 8;
        const int   col = mat * 32 + nt * 16 + (lane & 15);
#pragma unroll
        for (int r = 0; r < 8; ++r) {
            float val = c[r] + bn;
            if (mat != 2) val = fast_tanh(val);   // wave-uniform predicate
            s_cat[hi8 + r][col] = (_Float16)val;
        }
    }
    __syncthreads();

    // --- phase 2: gates = [obs|hx] @ [W_ih;W_hh]^T + (b_ih+b_hh) ---
    v16h fA[6];
#pragma unroll
    for (int ks = 0; ks < 6; ++ks)
        fA[ks] = load_A_frag<192>(&s_cat[0][0], lane, ks * 32);

#pragma unroll
    for (int ni = 0; ni < 3; ++ni) {
        const int nt = wave + ni * 8;                 // 0..23
        const int g0 = nt * 16;
        const float bn = b_ih[g0 + (lane & 15)] + b_hh[g0 + (lane & 15)];
        v8f c = {bn, bn, bn, bn, bn, bn, bn, bn};
#pragma unroll
        for (int ks = 0; ks < 6; ++ks) {
            // k-step never straddles the 96 boundary -> compile-time matrix pick
            const float* Wsrc = (ks < 3) ? W_ih : W_hh;
            const int    kb   = (ks < 3) ? ks * 32 : (ks - 3) * 32;
            v16h fb = load_B_frag_g2(Wsrc, H3, lane, g0, kb);
            c = wmma_f16(fA[ks], fb, c);
        }
        const int hi8 = (lane >> 4) * 8;
#pragma unroll
        for (int r = 0; r < 8; ++r)
            s_gates[hi8 + r][g0 + (lane & 15)] = c[r];
    }
    __syncthreads();

    // --- phase 3: LSTM elementwise (hardware tanh; sigmoid via tanh) ---
    for (int i = tid; i < 16 * H3; i += 256) {
        int m = i / H3, j = i % H3;
        int grow = row0 + m;
        float ii = fast_sigmoid(s_gates[m][j]);
        float ff = fast_sigmoid(s_gates[m][H3 + j]);
        float gg = fast_tanh   (s_gates[m][2 * H3 + j]);
        float oo = fast_sigmoid(s_gates[m][3 * H3 + j]);
        float cn = ff * cx[(size_t)grow * H3 + j] + ii * gg;
        float hn = oo * fast_tanh(cn);
        s_hnew[m][j] = hn;
        out[OFF_HX + (size_t)grow * H3 + j] = hn;
        out[OFF_CX + (size_t)grow * H3 + j] = cn;
    }
    __syncthreads();

    // --- phase 4: heads (96->4, 96->8) + argmax (first-max like jnp.argmax) ---
    if (tid < 192) {
        int m, n; const float* Wr; float acc;
        if (tid < 64) { m = tid >> 2; n = tid & 3; Wr = W_mov + n * H3; acc = b_mov[n]; }
        else { int t = tid - 64; m = t >> 3; n = t & 7; Wr = W_int + n * H3; acc = b_int[n]; }
#pragma unroll 8
        for (int k = 0; k < H3; ++k) acc += s_hnew[m][k] * Wr[k];
        if (tid < 64) { s_act[m][n] = acc;     out[(size_t)(row0 + m) * 4 + n] = acc; }
        else          { s_act[m][4 + n] = acc; out[OFF_AI + (size_t)(row0 + m) * 8 + n] = acc; }
    }
    __syncthreads();
    if (tid < 16) {
        const int m = tid;
        int   bi = 0;    float bv = s_act[m][0];
        for (int n = 1; n < 4; ++n) if (s_act[m][n] > bv) { bv = s_act[m][n]; bi = n; }
        out[OFF_M1 + row0 + m] = (float)bi;
        int   bj = 0;    float bw = s_act[m][4];
        for (int n = 1; n < 8; ++n) if (s_act[m][4 + n] > bw) { bw = s_act[m][4 + n]; bj = n; }
        out[OFF_M2 + row0 + m] = (float)bj;
    }
}

// ---------------------------------------------------------------------------
extern "C" void kernel_launch(void* const* d_in, const int* in_sizes, int n_in,
                              void* d_out, int out_size, void* d_ws, size_t ws_size,
                              hipStream_t stream) {
    const float* agents_obs = (const float*)d_in[0];   // [B,256,13]
    const float* lms_obs    = (const float*)d_in[1];   // [B,64,5]
    const float* c_obs      = (const float*)d_in[2];   // [B,18]
    const float* hx         = (const float*)d_in[3];   // [B,96]
    const float* cx         = (const float*)d_in[4];   // [B,96]
    const float* W_ag       = (const float*)d_in[5];
    const float* b_ag       = (const float*)d_in[6];
    const float* W_lm       = (const float*)d_in[7];
    const float* b_lm       = (const float*)d_in[8];
    const float* W_agg_ag   = (const float*)d_in[9];
    const float* b_agg_ag   = (const float*)d_in[10];
    const float* W_agg_lm   = (const float*)d_in[11];
    const float* b_agg_lm   = (const float*)d_in[12];
    const float* W_c        = (const float*)d_in[13];
    const float* b_c        = (const float*)d_in[14];
    const float* W_ih       = (const float*)d_in[15];
    const float* b_ih       = (const float*)d_in[16];
    const float* W_hh       = (const float*)d_in[17];
    const float* b_hh       = (const float*)d_in[18];
    const float* W_mov      = (const float*)d_in[19];
    const float* b_mov      = (const float*)d_in[20];
    const float* W_int      = (const float*)d_in[21];
    const float* b_int      = (const float*)d_in[22];
    (void)n_in; (void)out_size; (void)ws_size;

    const int B = in_sizes[3] / (3 * HID);             // 4096

    float* mean_a = (float*)d_ws;                      // [B,32]
    float* mean_l = mean_a + (size_t)B * HID;          // [B,32]

    encoder_mean_kernel<256, 13><<<B, 256, 0, stream>>>(agents_obs, W_ag, b_ag, mean_a);
    encoder_mean_kernel<64, 5><<<B, 64, 0, stream>>>(lms_obs, W_lm, b_lm, mean_l);
    head_kernel<<<B / 16, 256, 0, stream>>>(mean_a, mean_l, c_obs, hx, cx,
                                            W_agg_ag, b_agg_ag, W_agg_lm, b_agg_lm,
                                            W_c, b_c, W_ih, b_ih, W_hh, b_hh,
                                            W_mov, b_mov, W_int, b_int,
                                            (float*)d_out, B);
}
